// BasicFormerBlock_46531675685414
// MI455X (gfx1250) — matrix operand order repaired and checked by reading, not compiled, not run on hardware
//
#include <hip/hip_runtime.h>
#include <hip/hip_bf16.h>

typedef __bf16 bf16_t;
typedef __attribute__((ext_vector_type(16))) __bf16 v16bf;
typedef __attribute__((ext_vector_type(8)))  __bf16 v8bf;
typedef __attribute__((ext_vector_type(8)))  float  v8f;

union V16U { v16bf v; v8bf h[2]; };
union F4BF { float4 f; bf16_t e[8]; };

#if defined(__has_builtin)
#  if __has_builtin(__builtin_amdgcn_tensor_load_to_lds)
#    define USE_TDM 1
#  else
#    define USE_TDM 0
#  endif
#else
#  define USE_TDM 0
#endif

static __device__ __forceinline__ v8f wmma_bf16(v16bf a, v16bf b, v8f c) {
  // v_wmma_f32_16x16x32_bf16: D = A(16x32) x B(32x16) + C(16x16 f32)
  return __builtin_amdgcn_wmma_f32_16x16x32_bf16(false, a, false, b, (short)0, c,
                                                 false, false);
}

#if USE_TDM
typedef __attribute__((ext_vector_type(4))) unsigned tdm_v4u;
typedef __attribute__((ext_vector_type(8))) int      tdm_v8i;
typedef __attribute__((ext_vector_type(4))) int      tdm_v4i;

// 2D tile (tile_d1 rows x tile_d0 elements, 2-byte elements) global -> LDS,
// with LDS row padding: after 2^(padi+1) DWORDs insert (pada+1) DWORDs.
// D# layout per CDNA5 ISA 8.3/8.4 (group0 128b, group1 256b; groups 2/3 zero).
static __device__ __forceinline__ void tdm_load_2d_bf16(
    const void* gptr, unsigned lds_off, unsigned tensor_d0, unsigned tensor_d1,
    unsigned long long stride0, unsigned tile_d0, unsigned tile_d1,
    unsigned padi, unsigned pada) {
  unsigned long long ga = (unsigned long long)gptr;
  tdm_v4u g0;
  g0[0] = 1u;                                           // count=1 (valid user D#)
  g0[1] = lds_off;                                      // lds_addr [63:32]
  g0[2] = (unsigned)(ga & 0xffffffffu);                 // global_addr [95:64]
  g0[3] = (unsigned)((ga >> 32) & 0x01ffffffu)          // global_addr [120:96]
        | (2u << 30);                                   // type=2 ("image")
  unsigned w0 = (1u << 16)                              // data_size = 2 bytes
              | (1u << 20)                              // pad_enable
              | ((padi & 7u) << 22)                     // pad_interval
              | ((pada & 127u) << 25);                  // pad_amount
  unsigned w1 = (tensor_d0 & 0xffffu) << 16;            // tensor_dim0 @ bit48
  unsigned w2 = ((tensor_d0 >> 16) & 0xffffu)
              | ((tensor_d1 & 0xffffu) << 16);          // tensor_dim1 @ bit80
  unsigned w3 = ((tensor_d1 >> 16) & 0xffffu)
              | ((tile_d0 & 0xffffu) << 16);            // tile_dim0 @ bit112
  unsigned w4 = (tile_d1 & 0xffffu);                    // tile_dim1 @ bit128
  unsigned w5 = (unsigned)(stride0 & 0xffffffffu);      // dim0_stride @ bit160
  unsigned w6 = (unsigned)((stride0 >> 32) & 0xffffu);
  tdm_v8i g1;
  g1[0] = (int)w0; g1[1] = (int)w1; g1[2] = (int)w2; g1[3] = (int)w3;
  g1[4] = (int)w4; g1[5] = (int)w5; g1[6] = (int)w6; g1[7] = 0;
  tdm_v4i z4 = {0, 0, 0, 0};
#if __clang_major__ >= 23
  tdm_v8i z8 = {0, 0, 0, 0, 0, 0, 0, 0};
  __builtin_amdgcn_tensor_load_to_lds(g0, g1, z4, z4, z8, 0);
#else
  __builtin_amdgcn_tensor_load_to_lds(g0, g1, z4, z4, 0);
#endif
}
#endif  // USE_TDM

// ---------------------------------------------------------------- conversions
__global__ void f32_to_bf16_kernel(const float* __restrict__ in,
                                   bf16_t* __restrict__ out, long n) {
  long i = (long)blockIdx.x * blockDim.x + threadIdx.x;
  long stride = (long)gridDim.x * blockDim.x;
  for (; i < n; i += stride) out[i] = (bf16_t)in[i];
}

// ---------------------------------------------------------------- layer norm
__device__ __forceinline__ float block_sum256(float v, float* redbuf) {
#pragma unroll
  for (int off = 16; off > 0; off >>= 1) v += __shfl_xor(v, off, 32);
  if ((threadIdx.x & 31) == 0) redbuf[threadIdx.x >> 5] = v;
  __syncthreads();
  if (threadIdx.x == 0) {
    float t = 0.f;
#pragma unroll
    for (int i = 0; i < 8; ++i) t += redbuf[i];
    redbuf[0] = t;
  }
  __syncthreads();
  float t = redbuf[0];
  __syncthreads();
  return t;
}

__global__ void __launch_bounds__(256)
layernorm_bf16_kernel(const float* __restrict__ x, const float* __restrict__ g,
                      const float* __restrict__ bta, bf16_t* __restrict__ out,
                      int D) {
  __shared__ float redbuf[8];
  const int row = blockIdx.x;
  const float* xr = x + (long)row * D;
  const int base = threadIdx.x * 4;   // D == 1024, 256 threads
  float4 xv = *(const float4*)(xr + base);
  float v0 = xv.x, v1 = xv.y, v2 = xv.z, v3 = xv.w;
  float mean = block_sum256(v0 + v1 + v2 + v3, redbuf) * (1.0f / 1024.0f);
  float c0 = v0 - mean, c1 = v1 - mean, c2 = v2 - mean, c3 = v3 - mean;
  float var = block_sum256(c0*c0 + c1*c1 + c2*c2 + c3*c3, redbuf) * (1.0f / 1024.0f);
  float rstd = rsqrtf(var + 1e-12f);
  bf16_t* o = out + (long)row * D + base;
  const float* gp = g + base;
  const float* bp = bta + base;
  o[0] = (bf16_t)(c0 * rstd * gp[0] + bp[0]);
  o[1] = (bf16_t)(c1 * rstd * gp[1] + bp[1]);
  o[2] = (bf16_t)(c2 * rstd * gp[2] + bp[2]);
  o[3] = (bf16_t)(c3 * rstd * gp[3] + bp[3]);
}

// ---------------------------------------------------------------- GEMM (bf16 WMMA)
// C[M,N] = A[M,K]bf16 @ B[K,N]bf16  [+bias][SiLU][+res] -> f32 or bf16
// Block 128x128, BK=32, 256 threads = 8 waves, wave tile 32x64 (2x4 WMMA),
// double-buffered LDS (one barrier per K-step). A tile staged by TDM when
// available (pad 16 DWORDs -> +4 DWORDs reproduces the [.][40] layout).
template <bool HAS_BIAS, int ACT, bool HAS_RES, bool OUT_BF16>
__global__ void __launch_bounds__(256)
gemm_bf16_kernel(const bf16_t* __restrict__ A, const bf16_t* __restrict__ B,
                 const float* __restrict__ bias, const float* __restrict__ res,
                 float* __restrict__ Cf, bf16_t* __restrict__ Cb,
                 int M, int N, int K) {
  __shared__ bf16_t sA[2][128][40];    // [m][k], 80B stride
  __shared__ bf16_t sBt[2][128][40];   // [n][k] transposed

  const int tid = threadIdx.x;
  const int lane = tid & 31;
  const int wave = tid >> 5;
  const int wr = wave & 3;             // M (4 x 32)
  const int wc = wave >> 2;            // N (2 x 64)
  const int lrow = lane & 15;
  const int lhalf = lane >> 4;
  const int bm = blockIdx.y * 128;
  const int bn = blockIdx.x * 128;

  const int arow = tid >> 1;           // 0..127
  const int ak0  = (tid & 1) * 16;     // 0/16
  const int bk   = tid >> 3;           // 0..31
  const int bn0  = (tid & 7) * 16;     // 0..112

  v8f acc[2][4];
#pragma unroll
  for (int i = 0; i < 2; ++i)
#pragma unroll
    for (int j = 0; j < 4; ++j) acc[i][j] = {};

  auto stageA = [&](int kk, int bsel) {
#if USE_TDM
    if (wave == 0)
      tdm_load_2d_bf16(A + (size_t)bm * K + kk,
                       (unsigned)(unsigned long long)&sA[bsel][0][0],
                       (unsigned)K, (unsigned)M, (unsigned long long)K,
                       32u, 128u, /*padi 16DW*/3u, /*pada 4DW*/3u);
#else
    const bf16_t* ga = A + (long)(bm + arow) * K + kk + ak0;
    *(float4*)&sA[bsel][arow][ak0]     = *(const float4*)ga;
    *(float4*)&sA[bsel][arow][ak0 + 8] = *(const float4*)(ga + 8);
#endif
  };
  auto stageB = [&](int kk, int bsel) {
    const bf16_t* gb = B + (long)(kk + bk) * N + bn + bn0;
    F4BF u0; u0.f = *(const float4*)gb;
    F4BF u1; u1.f = *(const float4*)(gb + 8);
#pragma unroll
    for (int i = 0; i < 8; ++i) sBt[bsel][bn0 + i][bk]     = u0.e[i];
#pragma unroll
    for (int i = 0; i < 8; ++i) sBt[bsel][bn0 + 8 + i][bk] = u1.e[i];
  };

  stageA(0, 0);
  stageB(0, 0);
#if USE_TDM
  if (wave == 0) __builtin_amdgcn_s_wait_tensorcnt(0);
#endif
  __syncthreads();

  int buf = 0;
  for (int kk = 0; kk < K; kk += 32) {
    const int nxt = kk + 32;
    if (nxt < K) {
      stageA(nxt, buf ^ 1);
      stageB(nxt, buf ^ 1);
      if (nxt + 32 < K)
        __builtin_prefetch(B + (long)(nxt + 32 + bk) * N + bn + bn0, 0, 1);
    }

    v16bf am[2], bv[4];
#pragma unroll
    for (int i = 0; i < 2; ++i) {      // A frag: lane=M row, K = lhalf*8+e (+16)
      const int r = wr * 32 + i * 16 + lrow;
      const int k0 = lhalf * 8;
      V16U u;
      u.h[0] = *(const v8bf*)&sA[buf][r][k0];
      u.h[1] = *(const v8bf*)&sA[buf][r][k0 + 16];
      am[i] = u.v;
    }
#pragma unroll
    for (int j = 0; j < 4; ++j) {      // B frag: lane=N col, K = lhalf*8+e (+16)
      const int c = wc * 64 + j * 16 + lrow;
      const int k0 = lhalf * 8;
      V16U u;
      u.h[0] = *(const v8bf*)&sBt[buf][c][k0];
      u.h[1] = *(const v8bf*)&sBt[buf][c][k0 + 16];
      bv[j] = u.v;
    }
#pragma unroll
    for (int i = 0; i < 2; ++i)
#pragma unroll
      for (int j = 0; j < 4; ++j)
        acc[i][j] = wmma_bf16(am[i], bv[j], acc[i][j]);

#if USE_TDM
    if (wave == 0 && nxt < K) __builtin_amdgcn_s_wait_tensorcnt(0);
#endif
    __syncthreads();
    buf ^= 1;
  }

  // epilogue: C layout lane=N col, vgpr r -> M = r + lhalf*8
#pragma unroll
  for (int j = 0; j < 4; ++j) {
    const int gcol = bn + wc * 64 + j * 16 + lrow;
    float bval = 0.f;
    if (HAS_BIAS) bval = bias[gcol];
#pragma unroll
    for (int i = 0; i < 2; ++i) {
#pragma unroll
      for (int r = 0; r < 8; ++r) {
        const int grow = bm + wr * 32 + i * 16 + r + lhalf * 8;
        float v = acc[i][j][r] + bval;
        if (ACT == 1) v = v / (1.0f + __expf(-v));   // SiLU
        if (HAS_RES) v += res[(long)grow * N + gcol];
        if (OUT_BF16) Cb[(long)grow * N + gcol] = (bf16_t)v;
        else          Cf[(long)grow * N + gcol] = v;
      }
    }
  }
}

// ---------------------------------------------------------------- attention
// grid (S/128, H, B); 256 threads = 8 waves; wave owns 16 query rows.
__global__ void __launch_bounds__(256)
attention_kernel(const bf16_t* __restrict__ Q, const bf16_t* __restrict__ Km,
                 const bf16_t* __restrict__ Vm, const int* __restrict__ amask,
                 bf16_t* __restrict__ ctx, int S, int Hn) {
  __shared__ bf16_t sK[32][72];        // [key][feat], 144B stride
  __shared__ bf16_t sVt[64][40];       // [feat][key] (transposed)
  __shared__ bf16_t sP[8][16][40];     // per-wave P relayout slab

  const int b = blockIdx.z, h = blockIdx.y;
  const int qbase = blockIdx.x * 128;
  const int tid = threadIdx.x, lane = tid & 31, wave = tid >> 5;
  const int lrow = lane & 15, lhalf = lane >> 4;
  const int D = Hn * 64;
  const long headoff = (long)b * S * D + h * 64;
  const int qw = qbase + wave * 16;

  // Q fragments (A layout), contract halves over features 0-31 / 32-63
  v16bf qf[2];
  {
    const long qro = headoff + (long)(qw + lrow) * D;
#pragma unroll
    for (int c = 0; c < 2; ++c) {
      const int f0 = c * 32 + lhalf * 8;
      V16U u;
      u.h[0] = *(const v8bf*)(Q + qro + f0);
      u.h[1] = *(const v8bf*)(Q + qro + f0 + 16);
      qf[c] = u.v;
    }
  }

  v8f acc[4];
  acc[0] = {}; acc[1] = {}; acc[2] = {}; acc[3] = {};
  float mrow[8], lsum[8];
#pragma unroll
  for (int r = 0; r < 8; ++r) { mrow[r] = -3.0e38f; lsum[r] = 0.f; }

  const int nch = (qbase + 128) >> 5;            // causal: keys <= qbase+127
  const float SCL = 0.125f * 1.44269504f;        // (1/SCALE) * log2(e)

  for (int kc = 0; kc < nch; ++kc) {
    const int kbase = kc << 5;
    {   // stage K tile [32][64] (TDM if available) and V^T tile [64][32]
      const int key = tid >> 3;
      const int f0 = (tid & 7) * 8;
#if USE_TDM
      if (wave == 0)
        tdm_load_2d_bf16(Km + headoff + (long)kbase * D,
                         (unsigned)(unsigned long long)&sK[0][0],
                         64u, (unsigned)S, (unsigned long long)D,
                         64u, 32u, /*padi 32DW*/4u, /*pada 4DW*/3u);
#else
      const bf16_t* gk = Km + headoff + (long)(kbase + key) * D + f0;
      *(float4*)&sK[key][f0] = *(const float4*)gk;
#endif
      const bf16_t* gv = Vm + headoff + (long)(kbase + key) * D + f0;
      F4BF uv; uv.f = *(const float4*)gv;
#pragma unroll
      for (int i = 0; i < 8; ++i) sVt[f0 + i][key] = uv.e[i];
#if USE_TDM
      if (wave == 0) __builtin_amdgcn_s_wait_tensorcnt(0);
#endif
    }
    __syncthreads();

    if (kbase <= qw + 15) {                       // wave-uniform causal skip
      // scores: two 16x16 key sub-tiles, contract 64 = 2 x WMMA
      v8f sc[2];
#pragma unroll
      for (int jk = 0; jk < 2; ++jk) {
        v8f s = {};
#pragma unroll
        for (int c = 0; c < 2; ++c) {
          const bf16_t* kp = &sK[jk * 16 + lrow][c * 32 + lhalf * 8];
          V16U u; u.h[0] = *(const v8bf*)kp; u.h[1] = *(const v8bf*)(kp + 16);
          s = wmma_bf16(qf[c], u.v, s);
        }
        sc[jk] = s;
      }
      // scale + causal/pad mask, in log2 domain
      float sl[2][8];
#pragma unroll
      for (int jk = 0; jk < 2; ++jk) {
        const int key = kbase + jk * 16 + lrow;
        const int mk = amask[b * S + key];
#pragma unroll
        for (int r = 0; r < 8; ++r) {
          const int qg = qw + r + lhalf * 8;
          const bool ok = (key <= qg) && (mk != 0);
          sl[jk][r] = ok ? sc[jk][r] * SCL : -3.0e38f;
        }
      }
      // online softmax per row (row stats uniform across the 16-lane half)
#pragma unroll
      for (int r = 0; r < 8; ++r) {
        float cm = fmaxf(sl[0][r], sl[1][r]);
        cm = fmaxf(cm, __shfl_xor(cm, 1, 32));
        cm = fmaxf(cm, __shfl_xor(cm, 2, 32));
        cm = fmaxf(cm, __shfl_xor(cm, 4, 32));
        cm = fmaxf(cm, __shfl_xor(cm, 8, 32));
        const float mnew = fmaxf(mrow[r], cm);
        const float alpha = exp2f(mrow[r] - mnew);
        mrow[r] = mnew;
        const float p0 = exp2f(sl[0][r] - mnew);
        const float p1 = exp2f(sl[1][r] - mnew);
        float rs = p0 + p1;
        rs += __shfl_xor(rs, 1, 32);
        rs += __shfl_xor(rs, 2, 32);
        rs += __shfl_xor(rs, 4, 32);
        rs += __shfl_xor(rs, 8, 32);
        lsum[r] = lsum[r] * alpha + rs;
#pragma unroll
        for (int jf = 0; jf < 4; ++jf) acc[jf][r] *= alpha;
        const int rloc = r + lhalf * 8;
        sP[wave][rloc][lrow]      = (bf16_t)p0;   // C layout -> LDS
        sP[wave][rloc][16 + lrow] = (bf16_t)p1;
      }
      // reload P in A layout (same-wave DS ops are in-order)
      V16U up;
      {
        const bf16_t* pp = &sP[wave][lrow][lhalf * 8];
        up.h[0] = *(const v8bf*)pp;
        up.h[1] = *(const v8bf*)(pp + 16);
      }
      // ctx += P(16x32) @ V(32x64): 4 feature tiles
#pragma unroll
      for (int jf = 0; jf < 4; ++jf) {
        const bf16_t* vp = &sVt[jf * 16 + lrow][lhalf * 8];
        V16U uv2; uv2.h[0] = *(const v8bf*)vp; uv2.h[1] = *(const v8bf*)(vp + 16);
        acc[jf] = wmma_bf16(up.v, uv2.v, acc[jf]);
      }
    }
    __syncthreads();
  }

  float rl[8];
#pragma unroll
  for (int r = 0; r < 8; ++r) rl[r] = 1.0f / lsum[r];
#pragma unroll
  for (int jf = 0; jf < 4; ++jf) {
    const int col = jf * 16 + lrow;
#pragma unroll
    for (int r = 0; r < 8; ++r) {
      const int qg = qw + r + lhalf * 8;
      ctx[headoff + (long)qg * D + col] = (bf16_t)(acc[jf][r] * rl[r]);
    }
  }
}

// ---------------------------------------------------------------- launch
extern "C" void kernel_launch(void* const* d_in, const int* in_sizes, int n_in,
                              void* d_out, int out_size, void* d_ws, size_t ws_size,
                              hipStream_t stream) {
  (void)in_sizes; (void)n_in; (void)out_size; (void)ws_size;
  const int B = 4, S = 2048, D = 1024, F = 4096, Hn = 16;
  const int MS = B * S;  // 8192 rows

  const float* hidden = (const float*)d_in[0];
  const int*   amask  = (const int*)d_in[1];
  const float* Wq = (const float*)d_in[2];
  const float* Wk = (const float*)d_in[3];
  const float* Wv = (const float*)d_in[4];
  const float* Wo = (const float*)d_in[5];
  const float* ln1g = (const float*)d_in[6];
  const float* ln1b = (const float*)d_in[7];
  const float* W1 = (const float*)d_in[8];
  const float* b1 = (const float*)d_in[9];
  const float* W2 = (const float*)d_in[10];
  const float* b2 = (const float*)d_in[11];
  const float* ln2g = (const float*)d_in[12];
  const float* ln2b = (const float*)d_in[13];
  float* out = (float*)d_out;

  char* ws = (char*)d_ws;
  size_t off = 0;
  auto alloc = [&](size_t bytes) -> void* {
    void* p = ws + off;
    off = (off + bytes + 255) & ~(size_t)255;
    return p;
  };
  bf16_t* wqb  = (bf16_t*)alloc((size_t)D * D * 2);
  bf16_t* wkb  = (bf16_t*)alloc((size_t)D * D * 2);
  bf16_t* wvb  = (bf16_t*)alloc((size_t)D * D * 2);
  bf16_t* wob  = (bf16_t*)alloc((size_t)D * D * 2);
  bf16_t* w1b  = (bf16_t*)alloc((size_t)D * F * 2);
  bf16_t* w2b  = (bf16_t*)alloc((size_t)F * D * 2);
  bf16_t* xln1 = (bf16_t*)alloc((size_t)MS * D * 2);
  bf16_t* qb   = (bf16_t*)alloc((size_t)MS * D * 2);
  bf16_t* kb   = (bf16_t*)alloc((size_t)MS * D * 2);
  bf16_t* vb   = (bf16_t*)alloc((size_t)MS * D * 2);
  bf16_t* ctxb = (bf16_t*)alloc((size_t)MS * D * 2);
  float*  xres = (float*) alloc((size_t)MS * D * 4);
  bf16_t* yln2 = (bf16_t*)alloc((size_t)MS * D * 2);
  bf16_t* h1   = (bf16_t*)alloc((size_t)MS * F * 2);

  // weight conversions fp32 -> bf16
  f32_to_bf16_kernel<<<1024, 256, 0, stream>>>(Wq, wqb, (long)D * D);
  f32_to_bf16_kernel<<<1024, 256, 0, stream>>>(Wk, wkb, (long)D * D);
  f32_to_bf16_kernel<<<1024, 256, 0, stream>>>(Wv, wvb, (long)D * D);
  f32_to_bf16_kernel<<<1024, 256, 0, stream>>>(Wo, wob, (long)D * D);
  f32_to_bf16_kernel<<<1024, 256, 0, stream>>>(W1, w1b, (long)D * F);
  f32_to_bf16_kernel<<<1024, 256, 0, stream>>>(W2, w2b, (long)F * D);

  // LN1
  layernorm_bf16_kernel<<<MS, 256, 0, stream>>>(hidden, ln1g, ln1b, xln1, D);

  // QKV projections
  dim3 gD(D / 128, MS / 128);
  gemm_bf16_kernel<false, 0, false, true><<<gD, 256, 0, stream>>>(
      xln1, wqb, nullptr, nullptr, nullptr, qb, MS, D, D);
  gemm_bf16_kernel<false, 0, false, true><<<gD, 256, 0, stream>>>(
      xln1, wkb, nullptr, nullptr, nullptr, kb, MS, D, D);
  gemm_bf16_kernel<false, 0, false, true><<<gD, 256, 0, stream>>>(
      xln1, wvb, nullptr, nullptr, nullptr, vb, MS, D, D);

  // causal flash attention
  dim3 ga(S / 128, Hn, B);
  attention_kernel<<<ga, 256, 0, stream>>>(qb, kb, vb, amask, ctxb, S, Hn);

  // output projection + residual -> x (fp32)
  gemm_bf16_kernel<false, 0, true, false><<<gD, 256, 0, stream>>>(
      ctxb, wob, nullptr, hidden, xres, nullptr, MS, D, D);

  // LN2
  layernorm_bf16_kernel<<<MS, 256, 0, stream>>>(xres, ln2g, ln2b, yln2, D);

  // FFN: SiLU(y @ W1 + b1) @ W2 + b2 + x
  dim3 gF(F / 128, MS / 128);
  gemm_bf16_kernel<true, 1, false, true><<<gF, 256, 0, stream>>>(
      yln2, w1b, b1, nullptr, nullptr, h1, MS, F, D);
  gemm_bf16_kernel<true, 0, true, false><<<gD, 256, 0, stream>>>(
      h1, w2b, b2, xres, out, nullptr, MS, D, F);
}
